// Quantizer_module_55989193670842
// MI455X (gfx1250) — compile-verified, matrix operand order checked by reading, not run on hardware
//
#include <hip/hip_runtime.h>
#include <hip/hip_bf16.h>
#include <math.h>

// ---------------------------------------------------------------------------
// VQ quantizer for MI455X (gfx1250, wave32, WMMA, async global->LDS).
//   N = 131072 rows of x, N_E = 1024 codes, E_DIM = 256.
//   d_out layout (float): z_q [N*256] | min_indices [N] | loss [1]
//   d_ws  layout: wsq [1024 f32] @ 0 | weight_f16 [1024*256 f16] @ 4096
// ---------------------------------------------------------------------------

#define VQ_N     131072
#define VQ_NE    1024
#define VQ_ED    256
#define VQ_LSTR  264      // padded LDS column stride in halves (528 B)

typedef _Float16 v16h __attribute__((ext_vector_type(16)));
typedef _Float16 v8h  __attribute__((ext_vector_type(8)));
typedef float    v8f  __attribute__((ext_vector_type(8)));

// 128-bit transfer type for the async global->LDS builtin (param 1 is a
// global/AS1 pointer to a 16-byte int vector per the compiler diagnostic).
typedef int vq_v4i __attribute__((vector_size(16)));
typedef __attribute__((address_space(1))) vq_v4i* vq_gptr;
typedef __attribute__((address_space(3))) vq_v4i* vq_lptr;

#if __has_builtin(__builtin_amdgcn_global_load_async_to_lds_b128)
#define VQ_HAS_ASYNC 1
#else
#define VQ_HAS_ASYNC 0
#endif

__device__ __forceinline__ v16h vq_pack16(float4 f0, float4 f1, float4 f2, float4 f3) {
    v16h t;
    t[0]  = (_Float16)f0.x; t[1]  = (_Float16)f0.y; t[2]  = (_Float16)f0.z; t[3]  = (_Float16)f0.w;
    t[4]  = (_Float16)f1.x; t[5]  = (_Float16)f1.y; t[6]  = (_Float16)f1.z; t[7]  = (_Float16)f1.w;
    t[8]  = (_Float16)f2.x; t[9]  = (_Float16)f2.y; t[10] = (_Float16)f2.z; t[11] = (_Float16)f2.w;
    t[12] = (_Float16)f3.x; t[13] = (_Float16)f3.y; t[14] = (_Float16)f3.z; t[15] = (_Float16)f3.w;
    return t;
}

// Copy one 16-col x 256-K f16 weight chunk (8 KB) from global into a padded
// LDS buffer: 512 b128 transfers, exactly 2 async ops per thread.
__device__ __forceinline__ void vq_copy_chunk(const _Float16* __restrict__ gsrc,
                                              _Float16* lds_base, int t)
{
    #pragma unroll
    for (int e = t; e < 512; e += 256) {
        const int col  = e >> 5;     // 0..15
        const int part = e & 31;     // 0..31 (16B pieces of a 512B column)
        const char* s = (const char*)gsrc + col * 512 + part * 16;
        char*       d = (char*)lds_base + col * (VQ_LSTR * 2) + part * 16;
#if VQ_HAS_ASYNC
        __builtin_amdgcn_global_load_async_to_lds_b128(
            (vq_gptr)(char*)s, (vq_lptr)d, 0, 0);
#else
        *(float4*)d = *(const float4*)s;
#endif
    }
}

__device__ __forceinline__ void vq_wait_async0() {
#if VQ_HAS_ASYNC
    asm volatile("s_wait_asynccnt 0x0" ::: "memory");
#endif
}
// Async loads complete in order and each chunk is exactly 2 ops/thread, so
// "<= 2 outstanding" means every chunk but the newest has landed.
__device__ __forceinline__ void vq_wait_async2() {
#if VQ_HAS_ASYNC
    asm volatile("s_wait_asynccnt 0x2" ::: "memory");
#endif
}

// ---------------------------------------------------------------------------
// Kernel 1: weight f32 -> f16 copy, per-code squared norms, zero loss slot.
// ---------------------------------------------------------------------------
__global__ __launch_bounds__(256) void vq_prep_kernel(
    const float* __restrict__ w, float* __restrict__ wsq,
    _Float16* __restrict__ wf16, float* __restrict__ loss_slot)
{
    const int j = blockIdx.x;
    const int k = threadIdx.x;
    const float v = w[(size_t)j * VQ_ED + k];
    wf16[(size_t)j * VQ_ED + k] = (_Float16)v;

    __shared__ float red[256];
    red[k] = v * v;
    __syncthreads();
    #pragma unroll
    for (int s = 128; s > 0; s >>= 1) {
        if (k < s) red[k] += red[k + s];
        __syncthreads();
    }
    if (k == 0) {
        wsq[j] = red[0];
        if (j == 0) *loss_slot = 0.0f;
    }
}

// ---------------------------------------------------------------------------
// Kernel 2: argmin over codes via f16 WMMA, triple-buffered async global->LDS
// codebook staging. 1024 blocks x 256 threads; each wave owns 16 rows of x;
// all 8 waves share each LDS weight chunk.
// ---------------------------------------------------------------------------
__global__ __launch_bounds__(256) void vq_main_kernel(
    const float* __restrict__ x, const float* __restrict__ w,
    const _Float16* __restrict__ wf16, const float* __restrict__ wsq,
    float* __restrict__ zq, float* __restrict__ idx_out)
{
    const int lane = threadIdx.x & 31;
    const int wave = threadIdx.x >> 5;
    const int hi   = lane >> 4;      // 0: K {0-7,16-23}, 1: K {8-15,24-31}
    const int ln   = lane & 15;      // A: row within tile, B/C: column
    const int rowBase = blockIdx.x * 128 + wave * 16;

    __shared__ __align__(16) _Float16 wbuf[3][16 * VQ_LSTR];  // 3 x 8.25 KB
    __shared__ float swsq[VQ_NE];                             // 4 KB
    __shared__ int   sIdx[128];

    // Kick off async staging of chunks 0 and 1, then overlap the A-tile
    // conversion and wsq staging with the copies.
    vq_copy_chunk(wf16,                    wbuf[0], threadIdx.x);
    vq_copy_chunk(wf16 + (size_t)16 * VQ_ED, wbuf[1], threadIdx.x);

    for (int i = threadIdx.x; i < VQ_NE; i += 256) swsq[i] = wsq[i];

    // --- Load + convert the 16x256 A tile into 8 fragments (K-step = 32) ---
    v16h afrag[8];
    const float* xrow = x + (size_t)(rowBase + ln) * VQ_ED;
    #pragma unroll
    for (int ks = 0; ks < 8; ++ks) {
        const int k0 = ks * 32 + hi * 8;
        const float4 f0 = *(const float4*)(xrow + k0);
        const float4 f1 = *(const float4*)(xrow + k0 + 4);
        const float4 f2 = *(const float4*)(xrow + k0 + 16);
        const float4 f3 = *(const float4*)(xrow + k0 + 20);
        afrag[ks] = vq_pack16(f0, f1, f2, f3);
    }

    float bd[8];
    int   bj[8];
    #pragma unroll
    for (int v = 0; v < 8; ++v) { bd[v] = 3.4e38f; bj[v] = 0; }

    vq_wait_async2();      // chunk 0 landed (chunk 1 may still be in flight)
    __syncthreads();

    // --- Sweep codebook: 64 chunks of 16 columns, triple-buffered ---
    int rd = 0, wr = 2;
    for (int ci = 0; ci < 64; ++ci) {
        const int jb = ci << 4;

        // Stage chunk ci+2 while we compute on chunk ci (two chunks of
        // compute available to hide each copy).
        if (ci + 2 < 64)
            vq_copy_chunk(wf16 + (size_t)(jb + 32) * VQ_ED, wbuf[wr], threadIdx.x);

        const _Float16* wcol = &wbuf[rd][ln * VQ_LSTR];
        const float wq = swsq[jb + ln];

        // Preload the whole chunk's B fragments (one ds clause, one wait),
        // then run the 8 WMMAs back-to-back on two accumulator chains.
        v16h bfrag[8];
        #pragma unroll
        for (int ks = 0; ks < 8; ++ks) {
            const int k0 = ks * 32 + hi * 8;
            const v8h blo = *(const v8h*)(wcol + k0);        // ds_load_b128
            const v8h bhi = *(const v8h*)(wcol + k0 + 16);   // ds_load_b128
            bfrag[ks] = __builtin_shufflevector(blo, bhi,
                0, 1, 2, 3, 4, 5, 6, 7, 8, 9, 10, 11, 12, 13, 14, 15);
        }

        v8f c0 = {}, c1 = {};
        #pragma unroll
        for (int ks = 0; ks < 8; ks += 2) {
            c0 = __builtin_amdgcn_wmma_f32_16x16x32_f16(
                    false, afrag[ks],     false, bfrag[ks],     (short)0, c0, false, false);
            c1 = __builtin_amdgcn_wmma_f32_16x16x32_f16(
                    false, afrag[ks + 1], false, bfrag[ks + 1], (short)0, c1, false, false);
        }

        // d_j = ||w_j||^2 - 2 x.w_j  (||x||^2 is argmin-invariant)
        #pragma unroll
        for (int v = 0; v < 8; ++v) {
            const float d = fmaf(-2.0f, c0[v] + c1[v], wq);
            if (d < bd[v]) { bd[v] = d; bj[v] = jb + ln; }
        }

        if (ci < 62) vq_wait_async2();   // chunk ci+1 landed
        else         vq_wait_async0();   // tail: drain everything
        __syncthreads();

        rd = (rd == 2) ? 0 : rd + 1;
        wr = (wr == 2) ? 0 : wr + 1;
    }

    // --- Reduce across the 16 columns held in each 16-lane half ---
    #pragma unroll
    for (int v = 0; v < 8; ++v) {
        #pragma unroll
        for (int off = 8; off > 0; off >>= 1) {
            const float od = __shfl_xor(bd[v], off, 32);
            const int   oj = __shfl_xor(bj[v], off, 32);
            if (od < bd[v] || (od == bd[v] && oj < bj[v])) { bd[v] = od; bj[v] = oj; }
        }
    }

    // lanes 0 (rows 0-7) and 16 (rows 8-15) now hold per-row winners
    if (ln == 0) {
        #pragma unroll
        for (int v = 0; v < 8; ++v) {
            const int rloc = wave * 16 + hi * 8 + v;
            sIdx[rloc] = bj[v];
            idx_out[(size_t)blockIdx.x * 128 + rloc] = (float)bj[v];
        }
    }
    __syncthreads();

    // --- Gather z_q = weight[argmin] with full-wave float4 copies ---
    for (int r = 0; r < 16; ++r) {
        const int j = sIdx[wave * 16 + r];
        const float4* src = (const float4*)(w  + (size_t)j * VQ_ED);
        float4*       dst = (float4*)(zq + (size_t)(rowBase + r) * VQ_ED);
        dst[lane]      = src[lane];
        dst[lane + 32] = src[lane + 32];
    }
}

// ---------------------------------------------------------------------------
// Kernel 3: ce = -mean(diag(log_softmax(3 * W W^T))), gated by (idx == 0).
// ---------------------------------------------------------------------------
__global__ __launch_bounds__(256) void vq_loss_kernel(
    const float* __restrict__ w, const float* __restrict__ wsq,
    const int* __restrict__ idxp, float* __restrict__ loss_slot)
{
    const int i = blockIdx.x;
    const int t = threadIdx.x;

    __shared__ float wi[VQ_ED];
    wi[t] = w[(size_t)i * VQ_ED + t];
    __syncthreads();

    float l[4];
    float lmax = -3.4e38f;
    #pragma unroll
    for (int q = 0; q < 4; ++q) {
        const int j = t + q * 256;
        const float4* wj4 = (const float4*)(w + (size_t)j * VQ_ED);
        float dot = 0.0f;
        #pragma unroll 4
        for (int k4 = 0; k4 < VQ_ED / 4; ++k4) {
            const float4 a4 = wj4[k4];
            dot = fmaf(wi[4 * k4 + 0], a4.x, dot);
            dot = fmaf(wi[4 * k4 + 1], a4.y, dot);
            dot = fmaf(wi[4 * k4 + 2], a4.z, dot);
            dot = fmaf(wi[4 * k4 + 3], a4.w, dot);
        }
        l[q] = 3.0f * dot;
        lmax = fmaxf(lmax, l[q]);
    }

    __shared__ float red[256];
    red[t] = lmax;
    __syncthreads();
    #pragma unroll
    for (int s = 128; s > 0; s >>= 1) {
        if (t < s) red[t] = fmaxf(red[t], red[t + s]);
        __syncthreads();
    }
    const float mx = red[0];
    __syncthreads();

    float se = 0.0f;
    #pragma unroll
    for (int q = 0; q < 4; ++q) se += __expf(l[q] - mx);
    red[t] = se;
    __syncthreads();
    #pragma unroll
    for (int s = 128; s > 0; s >>= 1) {
        if (t < s) red[t] += red[t + s];
        __syncthreads();
    }

    if (t == 0) {
        const float lii  = 3.0f * wsq[i];               // diagonal logit
        const float logp = lii - mx - __logf(red[0]);
        const float gate = (*idxp == 0) ? 1.0f : 0.0f;
        atomicAdd(loss_slot, -logp * gate * (1.0f / (float)VQ_NE));
    }
}

// ---------------------------------------------------------------------------
extern "C" void kernel_launch(void* const* d_in, const int* in_sizes, int n_in,
                              void* d_out, int out_size, void* d_ws, size_t ws_size,
                              hipStream_t stream) {
    const float* x   = (const float*)d_in[0];   // [131072, 256] f32
    const float* w   = (const float*)d_in[1];   // [1024, 256] f32
    const int*   idx = (const int*)d_in[2];     // scalar

    float* zq       = (float*)d_out;
    float* idx_out  = zq + (size_t)VQ_N * VQ_ED;
    float* loss     = idx_out + VQ_N;

    float*    wsq  = (float*)d_ws;                          // 4 KB
    _Float16* wf16 = (_Float16*)((char*)d_ws + 4096);       // 512 KB

    vq_prep_kernel<<<VQ_NE, 256, 0, stream>>>(w, wsq, wf16, loss);
    vq_main_kernel<<<VQ_N / 128, 256, 0, stream>>>(x, w, wf16, wsq, zq, idx_out);
    vq_loss_kernel<<<VQ_NE, 256, 0, stream>>>(w, wsq, idx, loss);
}
